// Attention_28922309771646
// MI455X (gfx1250) — matrix operand order repaired and checked by reading, not compiled, not run on hardware
//
#include <hip/hip_runtime.h>
#include <hip/hip_bf16.h>

// ---------------- MLA dimensions ----------------
enum : int {
  D_MODEL = 2048, NH = 16, Q_LORA = 1536, KV_LORA = 512,
  ROPE = 64, NOPE = 128, VDIM = 128, QHD = 192, SEQ = 2048,
  MASK_W = SEQ + 150
};
#define ATTN_SCALE 0.08838834764831845f  /* 1/sqrt(128) */

typedef __attribute__((ext_vector_type(16))) __bf16 v16bf;
typedef __attribute__((ext_vector_type(8)))  float  v8f;

union Frag { unsigned int u[8]; v16bf v; };

// packed f32x2 -> bf16x2: single v_cvt_pk_bf16_f32 (CDNA packed convert)
__device__ inline unsigned int pack_bf(float lo, float hi) {
  unsigned int r;
  asm volatile("v_cvt_pk_bf16_f32 %0, %1, %2" : "=v"(r) : "v"(lo), "v"(hi));
  return r;
}
// A-matrix (16x32 bf16) K index for VGPR-pair p in lane-half h (ISA 7.12.2)
__device__ inline int amap_k(int p, int h) {
  return (p < 4) ? (8 * h + 2 * p) : (16 + 8 * h + 2 * (p - 4));
}
__device__ inline v8f wmma_bf16(const Frag& a, const Frag& b, v8f c) {
  return __builtin_amdgcn_wmma_f32_16x16x32_bf16(
      false, a.v, false, b.v, (short)0, c, false, false);
}
__device__ inline v8f v8f_zero() {
  v8f z;
  #pragma unroll
  for (int i = 0; i < 8; ++i) z[i] = 0.0f;
  return z;
}
__device__ inline void frag_from2(Frag& f, uint4 a, uint4 b) {
  f.u[0] = a.x; f.u[1] = a.y; f.u[2] = a.z; f.u[3] = a.w;
  f.u[4] = b.x; f.u[5] = b.y; f.u[6] = b.z; f.u[7] = b.w;
}
// A-fragment from packed-pair LDS row (pairs along K)
__device__ inline void load_afrag(Frag& f, const unsigned int* rowp, int hh) {
  uint4 lo = *(const uint4*)(rowp + 4 * hh);
  uint4 hi = *(const uint4*)(rowp + 8 + 4 * hh);
  frag_from2(f, lo, hi);
}
// B-fragment: packed pairs along K, table order [4h..4h+3] and [8+4h..8+4h+3]
__device__ inline void load_bfrag(Frag& f, const unsigned int* colp, int hh) {
  uint4 lo = *(const uint4*)(colp + 4 * hh);
  uint4 hi = *(const uint4*)(colp + 8 + 4 * hh);
  frag_from2(f, lo, hi);
}

// ================= generic fp32-in / bf16-WMMA / fp32-out GEMM =================
// C[M,N] = A[M,K] * B[K,N].  M,K multiples of 128/32 as launched; N guarded.
__global__ __launch_bounds__(256) void gemm_bf16_kernel(
    const float* __restrict__ A, const float* __restrict__ B,
    float* __restrict__ C, int M, int N, int K) {
  __shared__ unsigned int sAp[128][20];   // [row][k2] packed bf16 pairs (16 used)
  __shared__ unsigned int sBp[128][20];   // [col][k2] packed bf16 pairs (16 used)

  const int tid  = threadIdx.x;
  const int wave = tid >> 5, lane = tid & 31;
  const int r16  = lane & 15, hh = lane >> 4;
  const int wm   = wave >> 1, wn = wave & 1;      // 4x2 wave grid
  const int blockM = blockIdx.y * 128, blockN = blockIdx.x * 128;

  v8f acc[2][4];
  #pragma unroll
  for (int mt = 0; mt < 2; ++mt)
    #pragma unroll
    for (int nt = 0; nt < 4; ++nt) acc[mt][nt] = v8f_zero();

  for (int k0 = 0; k0 < K; k0 += 32) {
    __syncthreads();
    #pragma unroll
    for (int i = 0; i < 8; ++i) {            // A tile 128x32 -> 2048 packed dwords
      int e = i * 256 + tid;
      int r = e >> 4, c2 = e & 15;
      float2 v = *(const float2*)&A[(size_t)(blockM + r) * K + k0 + 2 * c2];
      sAp[r][c2] = pack_bf(v.x, v.y);
    }
    #pragma unroll
    for (int i = 0; i < 8; ++i) {            // B tile 32x128 -> 2048 packed dwords
      int e = i * 256 + tid;
      int col = e & 127, k2 = e >> 7;
      int gn = blockN + col;
      float b0 = 0.0f, b1 = 0.0f;
      if (gn < N) {
        b0 = B[(size_t)(k0 + 2 * k2) * N + gn];
        b1 = B[(size_t)(k0 + 2 * k2 + 1) * N + gn];
      }
      sBp[col][k2] = pack_bf(b0, b1);
    }
    if (k0 + 32 < K) {                       // prefetch next K tile
      if (tid < 128)
        __builtin_prefetch(&A[(size_t)(blockM + tid) * K + k0 + 32], 0, 0);
      int pr = tid >> 3, pc = (tid & 7) * 16;
      if (blockN + pc < N)
        __builtin_prefetch(&B[(size_t)(k0 + 32 + pr) * N + blockN + pc], 0, 0);
    }
    __syncthreads();

    Frag af[2], bfr[4];
    #pragma unroll
    for (int mt = 0; mt < 2; ++mt)
      load_afrag(af[mt], &sAp[wm * 32 + mt * 16 + r16][0], hh);
    #pragma unroll
    for (int nt = 0; nt < 4; ++nt)
      load_bfrag(bfr[nt], &sBp[wn * 64 + nt * 16 + r16][0], hh);
    #pragma unroll
    for (int mt = 0; mt < 2; ++mt)
      #pragma unroll
      for (int nt = 0; nt < 4; ++nt)
        acc[mt][nt] = wmma_bf16(af[mt], bfr[nt], acc[mt][nt]);
  }

  #pragma unroll
  for (int mt = 0; mt < 2; ++mt)
    #pragma unroll
    for (int nt = 0; nt < 4; ++nt)
      #pragma unroll
      for (int j = 0; j < 8; ++j) {
        int row = blockM + wm * 32 + mt * 16 + j + 8 * hh;  // C layout: VGPR j -> M=j+8h
        int col = blockN + wn * 64 + nt * 16 + r16;
        if (col < N) C[(size_t)row * N + col] = acc[mt][nt][j];
      }
}

// ================= row RMSNorm =================
__global__ __launch_bounds__(256) void rmsnorm_kernel(
    const float* __restrict__ in, const float* __restrict__ w,
    float* __restrict__ out, int width, int in_stride, int out_stride) {
  const int row = blockIdx.x;
  const float* x = in + (size_t)row * in_stride;
  float* y = out + (size_t)row * out_stride;
  float ss = 0.0f;
  for (int i = threadIdx.x; i < width; i += 256) { float v = x[i]; ss += v * v; }
  __shared__ float red[256];
  red[threadIdx.x] = ss; __syncthreads();
  for (int s = 128; s > 0; s >>= 1) {
    if (threadIdx.x < s) red[threadIdx.x] += red[threadIdx.x + s];
    __syncthreads();
  }
  float rs = rsqrtf(red[0] / (float)width + 1e-6f);
  for (int i = threadIdx.x; i < width; i += 256) y[i] = w[i] * x[i] * rs;
}

// ================= RoPE (in place on qproj pe slices + ckv k_pe) =================
__global__ void rope_kernel(float* __restrict__ qproj, float* __restrict__ ckv,
                            const float* __restrict__ freqs) {
  int idx = blockIdx.x * blockDim.x + threadIdx.x;
  int j = idx & 31; int rest = idx >> 5;
  int u = rest % (NH + 1); int s = rest / (NH + 1);
  if (s >= SEQ) return;
  float cs = freqs[(s * 32 + j) * 2 + 0];
  float sn = freqs[(s * 32 + j) * 2 + 1];
  float* p;
  if (u < NH) p = qproj + (size_t)s * (NH * QHD) + u * QHD + NOPE + 2 * j;
  else        p = ckv   + (size_t)s * (KV_LORA + ROPE) + KV_LORA + 2 * j;
  float a = p[0], b = p[1];
  p[0] = a * cs - b * sn;
  p[1] = a * sn + b * cs;
}

// ================= flash attention core (per head, 128-query blocks) =================
__global__ __launch_bounds__(256) void mla_attn_kernel(
    const float* __restrict__ qproj,   // [S, NH*192] (pe slices already roped)
    const float* __restrict__ ckv,     // [S, 576]    (cols 512..575 roped k_pe)
    const float* __restrict__ kvproj,  // [S, NH*256] (k_nope | v per head)
    const int*   __restrict__ mask,    // [S, S+150]
    float* __restrict__ attn) {        // [S, NH*128]
  const int h    = blockIdx.x;
  const int qblk = blockIdx.y;
  const int tid  = threadIdx.x;
  const int wave = tid >> 5, lane = tid & 31;
  const int r16  = lane & 15, hh = lane >> 4;

  __shared__ unsigned int sKp[32][100];      // [key][d2] packed pairs (96 used; 400B stride)
  __shared__ unsigned int sVp[128][20];      // [vd][k2] packed pairs (16 used)
  __shared__ int   sMask[128][32];
  __shared__ float sS[8][16][34];            // raw scores (C-layout scatter)
  __shared__ unsigned int sPp[8][16][20];    // packed P pairs (16 used)
  __shared__ float sM[8][16], sL[8][16], sF[8][16];

  const int qrow = qblk * 128 + wave * 16 + r16;

  // preload Q fragments (6 x 16x32 bf16), gathering pe|nope layout
  Frag qf[6];
  #pragma unroll
  for (int kk = 0; kk < 6; ++kk)
    #pragma unroll
    for (int p = 0; p < 8; ++p) {
      int d = kk * 32 + amap_k(p, hh);
      int col = (d < ROPE) ? (h * QHD + NOPE + d) : (h * QHD + (d - ROPE));
      float2 qv = *(const float2*)&qproj[(size_t)qrow * (NH * QHD) + col];
      qf[kk].u[p] = pack_bf(qv.x, qv.y);
    }

  v8f o[8];
  #pragma unroll
  for (int nt = 0; nt < 8; ++nt) o[nt] = v8f_zero();
  if (lane < 16) { sM[wave][lane] = -__builtin_inff(); sL[wave][lane] = 0.0f; }

  for (int kb = 0; kb < SEQ / 32; ++kb) {
    const int keybase = kb * 32;
    __syncthreads();
    // stage K block packed: sKp[t][d2] = pair(d=2*d2, d=2*d2+1)
    for (int e = tid; e < 32 * 96; e += 256) {
      int t = e / 96, d2 = e % 96;
      int key = keybase + t, d = 2 * d2;
      float2 kv2;
      if (d < ROPE)
        kv2 = *(const float2*)&ckv[(size_t)key * (KV_LORA + ROPE) + KV_LORA + d];
      else
        kv2 = *(const float2*)&kvproj[(size_t)key * (NH * (NOPE + VDIM))
                                      + h * (NOPE + VDIM) + (d - ROPE)];
      sKp[t][d2] = pack_bf(kv2.x, kv2.y);
    }
    // stage V block packed along keys: sVp[vd][k2] = pair(key=2k2, key=2k2+1)
    for (int e = tid; e < 128 * 16; e += 256) {
      int vd = e & 127, k2 = e >> 7;
      const float* vp = &kvproj[(size_t)(keybase + 2 * k2) * (NH * (NOPE + VDIM))
                                + h * (NOPE + VDIM) + NOPE + vd];
      sVp[vd][k2] = pack_bf(vp[0], vp[NH * (NOPE + VDIM)]);
    }
    // stage mask block (int2)
    for (int e = tid; e < 128 * 16; e += 256) {
      int qr = e >> 4, t2 = e & 15;
      int2 mv = *(const int2*)&mask[(size_t)(qblk * 128 + qr) * MASK_W
                                    + 150 + keybase + 2 * t2];
      sMask[qr][2 * t2] = mv.x; sMask[qr][2 * t2 + 1] = mv.y;
    }
    __syncthreads();

    // scores: two 16x16 tiles, K=192 in 6 chunks
    v8f sc[2]; sc[0] = v8f_zero(); sc[1] = v8f_zero();
    #pragma unroll
    for (int nt = 0; nt < 2; ++nt) {
      const unsigned int* keyrow = &sKp[nt * 16 + r16][0];
      #pragma unroll
      for (int kk = 0; kk < 6; ++kk) {
        Frag kb_;
        uint4 lo = *(const uint4*)(keyrow + kk * 16 + 4 * hh);
        uint4 hi = *(const uint4*)(keyrow + kk * 16 + 8 + 4 * hh);
        frag_from2(kb_, lo, hi);
        sc[nt] = wmma_bf16(qf[kk], kb_, sc[nt]);
      }
    }

    // scale + mask, scatter raw scores to per-wave LDS
    #pragma unroll
    for (int nt = 0; nt < 2; ++nt)
      #pragma unroll
      for (int j = 0; j < 8; ++j) {
        int rr = j + 8 * hh, cc = nt * 16 + r16;
        float s = sc[nt][j] * ATTN_SCALE;
        if (sMask[wave * 16 + rr][cc] == 1) s = -__builtin_inff();
        sS[wave][rr][cc] = s;
      }
    asm volatile("s_wait_dscnt 0" ::: "memory");

    // online softmax row update (one row per lane 0..15); emit packed bf16 P
    if (lane < 16) {
      int rr = lane;
      float mprev = sM[wave][rr], mmax = mprev;
      #pragma unroll
      for (int c = 0; c < 32; ++c) mmax = fmaxf(mmax, sS[wave][rr][c]);
      float f = (mmax == -__builtin_inff()) ? 1.0f : __expf(mprev - mmax);
      float lsum = 0.0f;
      #pragma unroll
      for (int c2 = 0; c2 < 16; ++c2) {
        float s0 = sS[wave][rr][2 * c2], s1 = sS[wave][rr][2 * c2 + 1];
        float e0 = (s0 == -__builtin_inff()) ? 0.0f : __expf(s0 - mmax);
        float e1 = (s1 == -__builtin_inff()) ? 0.0f : __expf(s1 - mmax);
        lsum += e0 + e1;
        sPp[wave][rr][c2] = pack_bf(e0, e1);
      }
      sM[wave][rr] = mmax;
      sL[wave][rr] = sL[wave][rr] * f + lsum;
      sF[wave][rr] = f;
    }
    asm volatile("s_wait_dscnt 0" ::: "memory");

    // rescale O accumulators
    float frow[8];
    #pragma unroll
    for (int j = 0; j < 8; ++j) frow[j] = sF[wave][j + 8 * hh];
    #pragma unroll
    for (int nt = 0; nt < 8; ++nt)
      #pragma unroll
      for (int j = 0; j < 8; ++j) o[nt][j] *= frow[j];

    // P (16x32) A-fragment, then PV into O
    Frag pf;
    load_afrag(pf, &sPp[wave][r16][0], hh);
    #pragma unroll
    for (int nt = 0; nt < 8; ++nt) {
      Frag vb;
      load_bfrag(vb, &sVp[nt * 16 + r16][0], hh);
      o[nt] = wmma_bf16(pf, vb, o[nt]);
    }
  }

  // normalize + store
  float linv[8];
  #pragma unroll
  for (int j = 0; j < 8; ++j) {
    float l = sL[wave][j + 8 * hh];
    linv[j] = (l > 0.0f) ? 1.0f / l : 0.0f;
  }
  #pragma unroll
  for (int nt = 0; nt < 8; ++nt)
    #pragma unroll
    for (int j = 0; j < 8; ++j) {
      int row = qblk * 128 + wave * 16 + j + 8 * hh;
      int col = h * VDIM + nt * 16 + r16;
      attn[(size_t)row * (NH * VDIM) + col] = o[nt][j] * linv[j];
    }
}

// ================= host-side pipeline =================
extern "C" void kernel_launch(void* const* d_in, const int* in_sizes, int n_in,
                              void* d_out, int out_size, void* d_ws, size_t ws_size,
                              hipStream_t stream) {
  (void)in_sizes; (void)n_in; (void)out_size; (void)ws_size;
  const float* x     = (const float*)d_in[0];
  const int*   mask  = (const int*)  d_in[1];
  const float* freqs = (const float*)d_in[2];
  const float* Wqa   = (const float*)d_in[3];
  const float* qln   = (const float*)d_in[4];
  const float* Wqb   = (const float*)d_in[5];
  const float* Wkva  = (const float*)d_in[6];
  const float* kvln  = (const float*)d_in[7];
  const float* Wkvb  = (const float*)d_in[8];
  const float* Wo    = (const float*)d_in[9];
  float* out = (float*)d_out;

  float* ws      = (float*)d_ws;
  float* q_lat   = ws;                                  // [2048,1536]
  float* qn      = q_lat + (size_t)SEQ * Q_LORA;        // [2048,1536]
  float* ckv     = qn + (size_t)SEQ * Q_LORA;           // [2048,576]
  float* compn   = ckv + (size_t)SEQ * (KV_LORA + ROPE);// [2048,512]
  float* qproj   = compn + (size_t)SEQ * KV_LORA;       // [2048,3072]
  float* kvproj  = qproj + (size_t)SEQ * (NH * QHD);    // [2048,4096]
  float* attnbuf = kvproj + (size_t)SEQ * (NH * (NOPE + VDIM)); // [2048,2048]

  dim3 blk(256);
  // q LoRA down + kv down-projection
  gemm_bf16_kernel<<<dim3(Q_LORA / 128, SEQ / 128), blk, 0, stream>>>(
      x, Wqa, q_lat, SEQ, Q_LORA, D_MODEL);
  gemm_bf16_kernel<<<dim3((KV_LORA + ROPE + 127) / 128, SEQ / 128), blk, 0, stream>>>(
      x, Wkva, ckv, SEQ, KV_LORA + ROPE, D_MODEL);
  // RMSNorms
  rmsnorm_kernel<<<SEQ, blk, 0, stream>>>(q_lat, qln, qn, Q_LORA, Q_LORA, Q_LORA);
  rmsnorm_kernel<<<SEQ, blk, 0, stream>>>(ckv, kvln, compn, KV_LORA, KV_LORA + ROPE, KV_LORA);
  // up-projections
  gemm_bf16_kernel<<<dim3(NH * QHD / 128, SEQ / 128), blk, 0, stream>>>(
      qn, Wqb, qproj, SEQ, NH * QHD, Q_LORA);
  gemm_bf16_kernel<<<dim3(NH * (NOPE + VDIM) / 128, SEQ / 128), blk, 0, stream>>>(
      compn, Wkvb, kvproj, SEQ, NH * (NOPE + VDIM), KV_LORA);
  // RoPE on q_pe slices + k_pe
  int rope_total = SEQ * (NH + 1) * 32;
  rope_kernel<<<(rope_total + 255) / 256, blk, 0, stream>>>(qproj, ckv, freqs);
  // attention core
  mla_attn_kernel<<<dim3(NH, SEQ / 128), blk, 0, stream>>>(
      qproj, ckv, kvproj, mask, attnbuf);
  // output projection
  gemm_bf16_kernel<<<dim3(D_MODEL / 128, SEQ / 128), blk, 0, stream>>>(
      attnbuf, Wo, out, SEQ, D_MODEL, NH * VDIM);
}
